// FeatureEncoder_41661182771388
// MI455X (gfx1250) — compile-verified, matrix operand order checked by reading, not run on hardware
//
#include <hip/hip_runtime.h>
#include <hip/hip_bf16.h>
#include <math.h>

typedef __attribute__((ext_vector_type(16))) _Float16 v16h;
typedef __attribute__((ext_vector_type(8)))  float    v8f;

// ---------------------------------------------------------------------------
// Fragment loaders (CDNA5 wave32 WMMA 16x16x32 f16 layouts, ISA 7.12.2)
//   A (16x32, M x K): lane L<16 -> row M=L,   K in {hi8+0..7, 16+hi8+0..7}, hi8=0
//                     lane L>=16-> row M=L-16, same pattern with hi8=8
//   B (32x16, K x N): identical pattern with N (weight output-channel row)
//                     in place of M, since W is stored [Cout][Cin] row-major.
// ---------------------------------------------------------------------------
template<int CIN, bool AFF>
__device__ __forceinline__ v16h load_a_frag(const float* __restrict__ Xrow, int kb,
                                            const float* __restrict__ sSc,
                                            const float* __restrict__ sSh) {
  v16h a;
  const int hi8 = (threadIdx.x & 16) ? 8 : 0;
#pragma unroll
  for (int e = 0; e < 16; ++e) {
    const int k = kb * 32 + (e < 8 ? hi8 + e : 16 + hi8 + (e - 8));
    float x = 0.0f;
    if (k < CIN) {
      x = Xrow[k];
      if (AFF) x = fmaxf(x * sSc[k] + sSh[k], 0.0f);   // fused BN-affine + ReLU
    }
    a[e] = (_Float16)x;
  }
  return a;
}

template<int CINP>
__device__ __forceinline__ v16h load_b_frag(const _Float16* __restrict__ sW,
                                            int nt, int kb) {
  v16h b;
  const int lane = threadIdx.x & 31;
  const int hi8  = (lane & 16) ? 8 : 0;
  const _Float16* p = sW + (nt * 16 + (lane & 15)) * CINP + kb * 32;
#pragma unroll
  for (int e = 0; e < 16; ++e) {
    const int k = (e < 8 ? hi8 + e : 16 + hi8 + (e - 8));
    b[e] = p[k];
  }
  return b;
}

// ---------------------------------------------------------------------------
// Store one 16x16 D tile (+ optional per-column sum/sumsq into LDS).
// GUARD=false is the fast path for full blocks: no per-row compares.
// D layout: VGPR r, lane L -> row (r + (L>>4)*8), col (L&15).
// ---------------------------------------------------------------------------
template<bool STATS, bool GUARD>
__device__ __forceinline__ void store_tile(const v8f& a, float* __restrict__ Y,
                                           float* __restrict__ sSum,
                                           float* __restrict__ sSq,
                                           int rowBase, int col, int stride,
                                           int hi, int Nrows) {
  float s1 = 0.f, s2 = 0.f;
#pragma unroll
  for (int r = 0; r < 8; ++r) {
    const int row = rowBase + r;
    const float y = a[r];
    if (!GUARD || row < Nrows) {
      Y[(size_t)row * stride + col] = y;
      s1 += y; s2 += y * y;
    }
  }
  if (STATS) {
    s1 += __shfl_xor(s1, 16, 32);   // combine the two half-lane row groups
    s2 += __shfl_xor(s2, 16, 32);
    if (hi == 0) { atomicAdd(sSum + col, s1); atomicAdd(sSq + col, s2); }
  }
}

// ---------------------------------------------------------------------------
// Generic fused kernel:  Y[N x COUT] = act(X * sc + sh) @ W^T
//   - 256 threads = 8 waves; 32 rows per wave as TWO independent 16-row
//     M-tiles -> two independent WMMA chains per wave fill the f16-WMMA
//     hazard slots and halve B-fragment LDS traffic. 256 rows per block.
//   - weights (f32->f16) and affine coefficients staged in LDS once per block
//   - optional per-column sum / sum-of-squares accumulation for BatchNorm
// ---------------------------------------------------------------------------
template<int CIN, int COUT, bool AFF, bool STATS>
__global__ __launch_bounds__(256)
void gemm_bn_kernel(const float* __restrict__ X, const float* __restrict__ W,
                    const float* __restrict__ sc, const float* __restrict__ sh,
                    float* __restrict__ Y, float* __restrict__ stats, int Nrows) {
  constexpr int CINP = (CIN < 32) ? 32 : CIN;   // pad K to a 32 multiple
  constexpr int KT   = CINP / 32;
  constexpr int NT   = COUT / 16;

  __shared__ _Float16 sW[COUT * CINP];
  __shared__ float    sStats[2 * COUT];
  __shared__ float    sSc[CINP];
  __shared__ float    sSh[CINP];

  const int tid = threadIdx.x;
  for (int i = tid; i < COUT * CINP; i += 256) {
    const int r = i / CINP, k = i % CINP;
    sW[i] = (k < CIN) ? (_Float16)W[r * CIN + k] : (_Float16)0.0f;
  }
  if (AFF) {
    for (int i = tid; i < CINP; i += 256) {
      sSc[i] = (i < CIN) ? sc[i] : 0.0f;
      sSh[i] = (i < CIN) ? sh[i] : 0.0f;
    }
  }
  if (STATS) {
    for (int i = tid; i < 2 * COUT; i += 256) sStats[i] = 0.0f;
  }
  __syncthreads();

  const int wave = tid >> 5;
  const int lane = tid & 31;
  const int blockRow = blockIdx.x * 256;
  const int rowA0 = blockRow + wave * 32 + (lane & 15);
  const int rowA1 = rowA0 + 16;
  const int rowA0c = rowA0 < Nrows ? rowA0 : (Nrows - 1);  // clamp: EXEC all-ones
  const int rowA1c = rowA1 < Nrows ? rowA1 : (Nrows - 1);
  const float* Xrow0 = X + (size_t)rowA0c * CIN;
  const float* Xrow1 = X + (size_t)rowA1c * CIN;

  v8f acc0[NT], acc1[NT];
#pragma unroll
  for (int nt = 0; nt < NT; ++nt) {
    acc0[nt] = (v8f){0.f, 0.f, 0.f, 0.f, 0.f, 0.f, 0.f, 0.f};
    acc1[nt] = acc0[nt];
  }

#pragma unroll
  for (int kb = 0; kb < KT; ++kb) {
    const v16h a0 = load_a_frag<CIN, AFF>(Xrow0, kb, sSc, sSh);
    const v16h a1 = load_a_frag<CIN, AFF>(Xrow1, kb, sSc, sSh);
#pragma unroll
    for (int nt = 0; nt < NT; ++nt) {
      const v16h b = load_b_frag<CINP>(sW, nt, kb);   // shared by both M-tiles
      acc0[nt] = __builtin_amdgcn_wmma_f32_16x16x32_f16(
          false, a0, false, b, (short)0, acc0[nt], false, false);
      acc1[nt] = __builtin_amdgcn_wmma_f32_16x16x32_f16(
          false, a1, false, b, (short)0, acc1[nt], false, false);
    }
  }

  const int hi   = lane >> 4;
  const int col0 = lane & 15;
  const int rowBase0 = blockRow + wave * 32 + hi * 8;
  const int rowBase1 = rowBase0 + 16;
  const bool full = (blockRow + 256 <= Nrows);  // block-uniform
  if (full) {
#pragma unroll
    for (int nt = 0; nt < NT; ++nt) {
      store_tile<STATS, false>(acc0[nt], Y, sStats, sStats + COUT,
                               rowBase0, nt * 16 + col0, COUT, hi, Nrows);
      store_tile<STATS, false>(acc1[nt], Y, sStats, sStats + COUT,
                               rowBase1, nt * 16 + col0, COUT, hi, Nrows);
    }
  } else {
#pragma unroll
    for (int nt = 0; nt < NT; ++nt) {
      store_tile<STATS, true>(acc0[nt], Y, sStats, sStats + COUT,
                              rowBase0, nt * 16 + col0, COUT, hi, Nrows);
      store_tile<STATS, true>(acc1[nt], Y, sStats, sStats + COUT,
                              rowBase1, nt * 16 + col0, COUT, hi, Nrows);
    }
  }
  if (STATS) {
    __syncthreads();
    for (int i = tid; i < 2 * COUT; i += 256) atomicAdd(&stats[i], sStats[i]);
  }
}

// ---------------------------------------------------------------------------
// Fused layer-1 for K and V towers: both read voxel_features, so compute
//   Y0 = X @ W0^T  and  Y1 = X @ W1^T   (CIN=64, COUT=16 each)
// in one pass -> one HBM read of X instead of two. 32 rows/wave, 4 WMMA
// chains per K step (2 M-tiles x 2 weight sets).
// ---------------------------------------------------------------------------
__global__ __launch_bounds__(256)
void gemm_dual16_kernel(const float* __restrict__ X,
                        const float* __restrict__ W0, const float* __restrict__ W1,
                        float* __restrict__ Y0, float* __restrict__ Y1,
                        float* __restrict__ stats0, float* __restrict__ stats1,
                        int Nrows) {
  constexpr int CIN = 64;
  __shared__ _Float16 sW[2 * 16 * CIN];
  __shared__ float    sStats[64];   // [sum0|sq0|sum1|sq1] x16

  const int tid = threadIdx.x;
  for (int i = tid; i < 16 * CIN; i += 256) {
    sW[i]            = (_Float16)W0[i];
    sW[16 * CIN + i] = (_Float16)W1[i];
  }
  for (int i = tid; i < 64; i += 256) sStats[i] = 0.0f;
  __syncthreads();

  const int wave = tid >> 5;
  const int lane = tid & 31;
  const int blockRow = blockIdx.x * 256;
  const int rowA0 = blockRow + wave * 32 + (lane & 15);
  const int rowA1 = rowA0 + 16;
  const int rowA0c = rowA0 < Nrows ? rowA0 : (Nrows - 1);
  const int rowA1c = rowA1 < Nrows ? rowA1 : (Nrows - 1);
  const float* Xrow0 = X + (size_t)rowA0c * CIN;
  const float* Xrow1 = X + (size_t)rowA1c * CIN;

  v8f z = (v8f){0.f, 0.f, 0.f, 0.f, 0.f, 0.f, 0.f, 0.f};
  v8f acc00 = z, acc01 = z, acc10 = z, acc11 = z;   // acc[tile][weightset]
#pragma unroll
  for (int kb = 0; kb < 2; ++kb) {
    const v16h a0 = load_a_frag<CIN, false>(Xrow0, kb, nullptr, nullptr);
    const v16h a1 = load_a_frag<CIN, false>(Xrow1, kb, nullptr, nullptr);
    const v16h b0 = load_b_frag<CIN>(sW, 0, kb);
    const v16h b1 = load_b_frag<CIN>(sW + 16 * CIN, 0, kb);
    acc00 = __builtin_amdgcn_wmma_f32_16x16x32_f16(false, a0, false, b0, (short)0, acc00, false, false);
    acc10 = __builtin_amdgcn_wmma_f32_16x16x32_f16(false, a1, false, b0, (short)0, acc10, false, false);
    acc01 = __builtin_amdgcn_wmma_f32_16x16x32_f16(false, a0, false, b1, (short)0, acc01, false, false);
    acc11 = __builtin_amdgcn_wmma_f32_16x16x32_f16(false, a1, false, b1, (short)0, acc11, false, false);
  }

  const int hi   = lane >> 4;
  const int col0 = lane & 15;
  const int rowBase0 = blockRow + wave * 32 + hi * 8;
  const int rowBase1 = rowBase0 + 16;
  const bool full = (blockRow + 256 <= Nrows);
  if (full) {
    store_tile<true, false>(acc00, Y0, sStats,      sStats + 16, rowBase0, col0, 16, hi, Nrows);
    store_tile<true, false>(acc10, Y0, sStats,      sStats + 16, rowBase1, col0, 16, hi, Nrows);
    store_tile<true, false>(acc01, Y1, sStats + 32, sStats + 48, rowBase0, col0, 16, hi, Nrows);
    store_tile<true, false>(acc11, Y1, sStats + 32, sStats + 48, rowBase1, col0, 16, hi, Nrows);
  } else {
    store_tile<true, true >(acc00, Y0, sStats,      sStats + 16, rowBase0, col0, 16, hi, Nrows);
    store_tile<true, true >(acc10, Y0, sStats,      sStats + 16, rowBase1, col0, 16, hi, Nrows);
    store_tile<true, true >(acc01, Y1, sStats + 32, sStats + 48, rowBase0, col0, 16, hi, Nrows);
    store_tile<true, true >(acc11, Y1, sStats + 32, sStats + 48, rowBase1, col0, 16, hi, Nrows);
  }
  __syncthreads();
  for (int i = tid; i < 32; i += 256) {
    atomicAdd(&stats0[i], sStats[i]);
    atomicAdd(&stats1[i], sStats[32 + i]);
  }
}

// ---------------------------------------------------------------------------
// BatchNorm stats -> per-column scale/shift:  s = g*rsqrt(var+eps), t = b - m*s
// ---------------------------------------------------------------------------
__global__ void bn_finalize_kernel(const float* __restrict__ stats,
                                   const float* __restrict__ g,
                                   const float* __restrict__ b,
                                   float* __restrict__ sc, float* __restrict__ sh,
                                   int C, float invN) {
  const int c = threadIdx.x;
  if (c < C) {
    const float m = stats[c] * invN;
    const float v = stats[C + c] * invN - m * m;   // biased variance
    const float s = g[c] * rsqrtf(v + 1e-3f);
    sc[c] = s;
    sh[c] = b[c] - m * s;
  }
}

__global__ void zero_ws_kernel(float* __restrict__ p, int n) {
  const int i = blockIdx.x * blockDim.x + threadIdx.x;
  if (i < n) p[i] = 0.0f;
}

// ---------------------------------------------------------------------------
// Per-sample 32x32 outer-product softmax attention. One wave per sample.
// Lane i owns output row i; k_j / v_j fetched via lane shuffles.
// max_j(q_i*k_j) = max(q_i*maxK, q_i*minK) -> one wave min/max pass shared by all i.
// ---------------------------------------------------------------------------
__global__ __launch_bounds__(256)
void attn_kernel(const float* __restrict__ Q, const float* __restrict__ K,
                 const float* __restrict__ V, float* __restrict__ F, int Nrows) {
  const int wave = threadIdx.x >> 5;
  const int lane = threadIdx.x & 31;
  const long n = (long)blockIdx.x * 8 + wave;
  if (n >= Nrows) return;                       // wave-uniform exit
  const float invs = 0.17677669529663687f;      // 1/sqrt(32)
  const float q = Q[n * 32 + lane];
  const float k = K[n * 32 + lane];
  const float v = V[n * 32 + lane];
  float kmax = k, kmin = k;
#pragma unroll
  for (int o = 16; o > 0; o >>= 1) {
    kmax = fmaxf(kmax, __shfl_xor(kmax, o, 32));
    kmin = fminf(kmin, __shfl_xor(kmin, o, 32));
  }
  const float m = fmaxf(q * kmax, q * kmin) * invs;
  float se = 0.f, sv = 0.f;
#pragma unroll
  for (int j = 0; j < 32; ++j) {
    const float kj = __shfl(k, j, 32);
    const float vj = __shfl(v, j, 32);
    const float e = __expf(q * kj * invs - m);
    se += e; sv += e * vj;
  }
  F[n * 32 + lane] = sv / se;
}

// ---------------------------------------------------------------------------
// Host side
// ---------------------------------------------------------------------------
struct TowerP { const float *w1, *g1, *b1, *w2, *g2, *b2, *w3; };

static inline TowerP get_tower(void* const* d_in, int base, bool sorted) {
  TowerP t;
  auto f = [&](int k) { return (const float*)d_in[base + k]; };
  if (!sorted) { t.w1=f(0); t.g1=f(1); t.b1=f(2); t.w2=f(3); t.g2=f(4); t.b2=f(5); t.w3=f(6); }
  else         { t.b1=f(0); t.b2=f(1); t.g1=f(2); t.g2=f(3); t.w1=f(4); t.w2=f(5); t.w3=f(6); }
  return t;
}

extern "C" void kernel_launch(void* const* d_in, const int* in_sizes, int n_in,
                              void* d_out, int out_size, void* d_ws, size_t ws_size,
                              hipStream_t stream) {
  (void)n_in; (void)out_size; (void)ws_size;
  const float* VF = (const float*)d_in[0];   // voxel_features [N,64]
  const float* NF = (const float*)d_in[1];   // norm_features  [N,64]
  const int N = in_sizes[0] / 64;

  // Param flattening order: dict insertion order (w1 first -> size 1024)
  // vs jax-sorted tree order (k-tower b1 first -> size 16).
  const bool sorted = (in_sizes[4] != 1024);
  TowerP tq, tk, tv, to;
  if (!sorted) {
    tq = get_tower(d_in,  4, false);
    tk = get_tower(d_in, 11, false);
    tv = get_tower(d_in, 18, false);
    to = get_tower(d_in, 25, false);
  } else {  // sorted top-level keys: k, out, q, scale, v
    tk = get_tower(d_in,  4, true);
    to = get_tower(d_in, 11, true);
    tq = get_tower(d_in, 18, true);
    tv = get_tower(d_in, 26, true);
  }

  float* ws    = (float*)d_ws;
  float* stats = ws;                     // 8 stages x 64 floats (sum | sumsq)
  float* ssbuf = ws + 512;               // 8 stages x 64 floats (scale | shift)
  float* bufA  = ws + 1024;
  const size_t stride = (size_t)N * 32;
  float* bufB = bufA + stride;
  float* Qb   = bufB + stride;
  float* Kb   = Qb   + stride;
  float* Vb   = Kb   + stride;

  const dim3 blk(256);
  const dim3 grid((N + 255) / 256);
  const float invN = 1.0f / (float)N;
  auto SS = [&](int s) { return ssbuf + s * 64; };
  auto ST = [&](int s) { return stats + s * 64; };

  zero_ws_kernel<<<dim3(2), blk, 0, stream>>>(stats, 512);

  // ---- Q tower (norm_features) ----
  gemm_bn_kernel<64,16,false,true ><<<grid, blk, 0, stream>>>(NF,   tq.w1, nullptr, nullptr, bufA, ST(0), N);
  bn_finalize_kernel<<<1, 64, 0, stream>>>(ST(0), tq.g1, tq.b1, SS(0), SS(0)+32, 16, invN);
  gemm_bn_kernel<16,32,true ,true ><<<grid, blk, 0, stream>>>(bufA, tq.w2, SS(0), SS(0)+32, bufB, ST(1), N);
  bn_finalize_kernel<<<1, 64, 0, stream>>>(ST(1), tq.g2, tq.b2, SS(1), SS(1)+32, 32, invN);
  gemm_bn_kernel<32,32,true ,false><<<grid, blk, 0, stream>>>(bufB, tq.w3, SS(1), SS(1)+32, Qb, nullptr, N);

  // ---- K + V tower layer 1, fused: single read of voxel_features ----
  //      Y1k -> bufA, Y1v -> Vb (first N*16 floats; Vb is rewritten by L3v later)
  gemm_dual16_kernel<<<grid, blk, 0, stream>>>(VF, tk.w1, tv.w1, bufA, Vb, ST(2), ST(4), N);

  // ---- K tower layers 2-3 ----
  bn_finalize_kernel<<<1, 64, 0, stream>>>(ST(2), tk.g1, tk.b1, SS(2), SS(2)+32, 16, invN);
  gemm_bn_kernel<16,32,true ,true ><<<grid, blk, 0, stream>>>(bufA, tk.w2, SS(2), SS(2)+32, bufB, ST(3), N);
  bn_finalize_kernel<<<1, 64, 0, stream>>>(ST(3), tk.g2, tk.b2, SS(3), SS(3)+32, 32, invN);
  gemm_bn_kernel<32,32,true ,false><<<grid, blk, 0, stream>>>(bufB, tk.w3, SS(3), SS(3)+32, Kb, nullptr, N);

  // ---- V tower layers 2-3 ----
  bn_finalize_kernel<<<1, 64, 0, stream>>>(ST(4), tv.g1, tv.b1, SS(4), SS(4)+32, 16, invN);
  gemm_bn_kernel<16,32,true ,true ><<<grid, blk, 0, stream>>>(Vb,   tv.w2, SS(4), SS(4)+32, bufB, ST(5), N);
  bn_finalize_kernel<<<1, 64, 0, stream>>>(ST(5), tv.g2, tv.b2, SS(5), SS(5)+32, 32, invN);
  gemm_bn_kernel<32,32,true ,false><<<grid, blk, 0, stream>>>(bufB, tv.w3, SS(5), SS(5)+32, Vb, nullptr, N);

  // ---- attention (writes F into bufA) ----
  attn_kernel<<<dim3((N + 7) / 8), blk, 0, stream>>>(Qb, Kb, Vb, bufA, N);

  // ---- output tower ----
  gemm_bn_kernel<32,32,false,true ><<<grid, blk, 0, stream>>>(bufA, to.w1, nullptr, nullptr, bufB, ST(6), N);
  bn_finalize_kernel<<<1, 64, 0, stream>>>(ST(6), to.g1, to.b1, SS(6), SS(6)+32, 32, invN);
  gemm_bn_kernel<32,16,true ,true ><<<grid, blk, 0, stream>>>(bufB, to.w2, SS(6), SS(6)+32, bufA, ST(7), N);
  bn_finalize_kernel<<<1, 64, 0, stream>>>(ST(7), to.g2, to.b2, SS(7), SS(7)+32, 16, invN);
  gemm_bn_kernel<16,64,true ,false><<<grid, blk, 0, stream>>>(bufA, to.w3, SS(7), SS(7)+32, (float*)d_out, nullptr, N);
}
